// PhaseUPDeT3_36593121361957
// MI455X (gfx1250) — matrix operand order repaired and checked by reading, not compiled, not run on hardware
//
#include <hip/hip_runtime.h>
#include <hip/hip_bf16.h>

typedef __attribute__((ext_vector_type(16))) _Float16 v16h;
typedef __attribute__((ext_vector_type(8)))  _Float16 v8h;
typedef __attribute__((ext_vector_type(8)))  float    v8f;

namespace {

constexpr int TOK   = 32;   // token dim
constexpr int TIN   = 32;   // input tokens
constexpr int EMBD  = 64;
constexpr int HEADS = 4;
constexpr int NDEP  = 2;
constexpr int TSEQ  = 33;   // tokens + hidden state
constexpr int TP    = 48;   // padded sequence rows
constexpr int FFD   = 256;  // 4*EMB == HEADS*EMB
constexpr int QOUT  = 26;   // 6 + 20 enemies

// ---- workspace layout (in halfs): f16 weights pre-swizzled to B-fragment order ----
constexpr size_t SZ_BIG  = 32 * 512;              // 64x256 or 256x64 => 32 tiles
constexpr size_t OFF_EMB = 0;                     // 32x64  => 4 tiles
constexpr size_t OFF_WQ  = OFF_EMB + 4 * 512;
constexpr size_t OFF_WK  = OFF_WQ + 2 * SZ_BIG;
constexpr size_t OFF_WV  = OFF_WK + 2 * SZ_BIG;
constexpr size_t OFF_WU  = OFF_WV + 2 * SZ_BIG;
constexpr size_t OFF_W1  = OFF_WU + 2 * SZ_BIG;
constexpr size_t OFF_W2  = OFF_W1 + 2 * SZ_BIG;
constexpr size_t OFF_TOP = OFF_W2 + 2 * SZ_BIG;   // 64x64 => 8 tiles

// ---------------- fragment index math (ISA 7.12.2, wave32) ----------------
// A fragment (16x32): lanes 0-15 -> M=lane, K in {0..7,16..23}; lanes 16-31 -> K in {8..15,24..31}
__device__ __forceinline__ int afrag_idx(int m, int k, int ktiles) {
  const int mt = m >> 4, mm = m & 15;
  const int kt = k >> 5, kk = k & 31;
  const int sel = (kk >> 3) & 1;
  const int lane = mm + (sel << 4);
  const int kk0 = kk - (sel << 3);
  const int h = (kk0 < 8) ? kk0 : (kk0 - 8);
  return ((mt * ktiles + kt) << 9) + (lane << 4) + h;
}

// A-fragment store base for a C-tile: element r lives at base + (r<<4)
__device__ __forceinline__ int afrag_base(int mb, int k, int ktiles) {
  const int kt = k >> 5, kk = k & 31;
  const int sel = (kk >> 3) & 1;
  const int kk0 = kk - (sel << 3);
  const int h = (kk0 < 8) ? kk0 : (kk0 - 8);
  return (((mb >> 4) * ktiles + kt) << 9) + (((mb & 15) + (sel << 4)) << 4) + h;
}

// B fragment (32x16): lane = (n&15) + 16*((k>>3)&1), half index = (k&7) + 8*((k>>4)&1)
__device__ __forceinline__ int bfrag_idx(int k, int n, int ntiles) {
  const int tile = (k >> 5) * ntiles + (n >> 4);
  const int lane = (n & 15) + (((k >> 3) & 1) << 4);
  return (tile << 9) + (lane << 4) + (k & 7) + (((k >> 4) & 1) << 3);
}

// One fragment = 512 halfs; lane L owns 16 contiguous halfs (32B -> 2x b128 loads)
__device__ __forceinline__ v16h load_frag(const _Float16* buf, int tile) {
  return *reinterpret_cast<const v16h*>(buf + (tile << 9) + ((threadIdx.x & 31) << 4));
}

__device__ __forceinline__ v8f wmma16(v16h a, v16h b, v8f c) {
  return __builtin_amdgcn_wmma_f32_16x16x32_f16(false, a, false, b, (short)0, c,
                                                false, false);
}

// A fragment gathered from f32 global (embedding inputs only)
__device__ __forceinline__ v16h load_a_gf32(const float* src, int ld, int m0, int k0) {
  const int L  = threadIdx.x & 31;
  const int m  = m0 + (L & 15);
  const int kb = (L < 16) ? 0 : 8;
  v16h a;
#pragma unroll
  for (int h = 0; h < 16; ++h) {
    const int v = h >> 1, p = h & 1;
    const int k = k0 + ((v < 4) ? (kb + 2 * v + p) : (16 + kb + 2 * (v - 4) + p));
    a[h] = (_Float16)src[m * ld + k];
  }
  return a;
}

// ---------------- C/D stores (lane L -> N=n0+(L&15); vgpr r -> M=m0+r+(L<16?0:8)) ----------------
__device__ __forceinline__ void store_c_f32(const v8f& c, float* dst, int ld,
                                            int m0, int n0) {
  const int L = threadIdx.x & 31;
  const int n = n0 + (L & 15);
  const int mb = m0 + ((L < 16) ? 0 : 8);
#pragma unroll
  for (int r = 0; r < 8; ++r) dst[(mb + r) * ld + n] = c[r];
}

__device__ __forceinline__ void store_c_f32_bias(const v8f& c, float* dst, int ld,
                                                 int m0, int n0, const float* bias) {
  const int L = threadIdx.x & 31;
  const int n = n0 + (L & 15);
  const int mb = m0 + ((L < 16) ? 0 : 8);
  const float bv = bias[n];
#pragma unroll
  for (int r = 0; r < 8; ++r) dst[(mb + r) * ld + n] = c[r] + bv;
}

__device__ __forceinline__ void store_c_f32_res(const v8f& c, float* dst, int ld,
                                                int m0, int n0, const float* bias,
                                                const float* res) {
  const int L = threadIdx.x & 31;
  const int n = n0 + (L & 15);
  const int mb = m0 + ((L < 16) ? 0 : 8);
  const float bv = bias[n];
#pragma unroll
  for (int r = 0; r < 8; ++r)
    dst[(mb + r) * ld + n] = c[r] + bv + res[(mb + r) * ld + n];
}

// store straight into an A-fragment-layout f16 buffer (col n == next GEMM's K index)
__device__ __forceinline__ void store_c_afrag(const v8f& c, _Float16* dst, int ktiles,
                                              int m0, int n0) {
  const int L = threadIdx.x & 31;
  const int n = n0 + (L & 15);
  const int mb = m0 + ((L < 16) ? 0 : 8);
  const int base = afrag_base(mb, n, ktiles);
#pragma unroll
  for (int r = 0; r < 8; ++r) dst[base + (r << 4)] = (_Float16)c[r];
}

__device__ __forceinline__ void store_c_afrag_relu(const v8f& c, _Float16* dst, int ktiles,
                                                   int m0, int n0, const float* bias) {
  const int L = threadIdx.x & 31;
  const int n = n0 + (L & 15);
  const int mb = m0 + ((L < 16) ? 0 : 8);
  const float bv = bias[n];
  const int base = afrag_base(mb, n, ktiles);
#pragma unroll
  for (int r = 0; r < 8; ++r)
    dst[base + (r << 4)] = (_Float16)fmaxf(c[r] + bv, 0.0f);
}

// embedding: write f32 row-major (for residual/LN) AND f16 A-fragment copy
__device__ __forceinline__ void store_c_emb(const v8f& c, float* xb, _Float16* xf,
                                            int m0, int n0, const float* bias) {
  const int L = threadIdx.x & 31;
  const int n = n0 + (L & 15);
  const int mb = m0 + ((L < 16) ? 0 : 8);
  const float bv = bias[n];
  const int base = afrag_base(mb, n, 2);
#pragma unroll
  for (int r = 0; r < 8; ++r) {
    const float val = c[r] + bv;
    xb[(mb + r) * EMBD + n] = val;
    xf[base + (r << 4)] = (_Float16)val;
  }
}

// q -> per-head A-fragments; k -> per-head transposed B-fragments; v -> per-head B-fragments
// All three store patterns are linear in r: q/k stride 16 halfs, v stride 1 (vectorized b128).
__device__ __forceinline__ void store_qkv(const v8f& c, _Float16* qf, _Float16* kf,
                                          _Float16* vf, int which, int m0, int n0,
                                          float mul) {
  const int L = threadIdx.x & 31;
  const int n = n0 + (L & 15);
  const int hh = n >> 6, cc = n & 63;
  const int mb = m0 + ((L < 16) ? 0 : 8);
  if (which == 0) {            // A: M=t, K=e (per head, ktiles=2)
    const int base = hh * 3072 + afrag_base(mb, cc, 2);
#pragma unroll
    for (int r = 0; r < 8; ++r) qf[base + (r << 4)] = (_Float16)(c[r] * mul);
  } else if (which == 1) {     // B^T: K=e (const), N=t (varies) -> stride 16
    const int base = hh * 3072 + bfrag_idx(cc, mb, 3);
#pragma unroll
    for (int r = 0; r < 8; ++r) kf[base + (r << 4)] = (_Float16)(c[r] * mul);
  } else {                     // B: K=t (varies), N=e (const) -> stride 1, 16B aligned
    const int base = hh * 4096 + bfrag_idx(mb, cc, 4);
    v8h pk;
#pragma unroll
    for (int r = 0; r < 8; ++r) pk[r] = (_Float16)c[r];
    *reinterpret_cast<v8h*>(vf + base) = pk;
  }
}

// ---------------- wave32 reductions ----------------
__device__ __forceinline__ float wave_sum(float v) {
#pragma unroll
  for (int o = 16; o > 0; o >>= 1) v += __shfl_xor(v, o, 32);
  return v;
}
__device__ __forceinline__ float wave_max(float v) {
#pragma unroll
  for (int o = 16; o > 0; o >>= 1) v = fmaxf(v, __shfl_xor(v, o, 32));
  return v;
}

}  // namespace

// ---- prep: f32 row-major weight -> f16 B-fragment-swizzled in workspace ----
__global__ void swizzle_weight(const float* __restrict__ src, _Float16* __restrict__ dst,
                               int K, int N) {
  const int idx = blockIdx.x * 256 + threadIdx.x;
  if (idx >= K * N) return;
  const int k = idx / N, n = idx - k * N;
  dst[bfrag_idx(k, n, N >> 4)] = (_Float16)src[idx];
}

__global__ __launch_bounds__(256)
void phase_updet_wmma(
    const float* __restrict__ inputs, const float* __restrict__ hidden,
    const float* __restrict__ p_h, const float* __restrict__ emb_b,
    const float* __restrict__ bu,
    const float* __restrict__ ln1_g, const float* __restrict__ ln1_b,
    const float* __restrict__ ffb1, const float* __restrict__ ffb2,
    const float* __restrict__ ln2_g, const float* __restrict__ ln2_b,
    const float* __restrict__ topb,
    const float* __restrict__ cW1, const float* __restrict__ cb1,
    const float* __restrict__ cW2, const float* __restrict__ cb2,
    const float* __restrict__ cW3, const float* __restrict__ cb3,
    const float* __restrict__ qsW, const float* __restrict__ qsb,
    const float* __restrict__ qiW, const float* __restrict__ qib,
    const float* __restrict__ pew, const _Float16* __restrict__ ws,
    float* __restrict__ out, int Bn) {
  __shared__ __align__(32) float    xbuf[TP * EMBD];   // x, f32 row-major (rows>=33 stay 0)
  __shared__ __align__(32) float    sbuf[TP * EMBD];   // scores / pre-LN / outputs (+ scratch tail)
  __shared__ __align__(32) _Float16 xfrag[6 * 512];    // x as A-fragments  (M=48,K=64)
  __shared__ __align__(32) _Float16 qfrag[24 * 512];   // q per head, A-fragments
  __shared__ __align__(32) _Float16 kfrag[24 * 512];   // k^T per head, B-fragments
  __shared__ __align__(32) _Float16 vfrag[32 * 512];   // v per head, B-fragments (K padded to 64)
  __shared__ __align__(32) _Float16 ofrag[24 * 512];   // attn-out / ff-hidden, A-fragments (K=256)
  __shared__ __align__(32) _Float16 afrag[6 * 512];    // softmax(att), A-fragments (K padded to 64)

  const int tid  = threadIdx.x;
  const int wave = tid >> 5;
  const int lane = tid & 31;
  const int b    = blockIdx.x;
  const float scale = 0.35355339059327373f;  // 64^-0.25

  // ---- init: zero buffers whose padding must read as exact zeros ----
  for (int i = tid; i < TP * EMBD; i += 256) xbuf[i] = 0.0f;
  for (int i = tid; i < 6 * 512; i += 256) xfrag[i] = (_Float16)0.0f;
  for (int i = tid; i < 6 * 512; i += 256) afrag[i] = (_Float16)0.0f;
  for (int i = tid; i < 32 * 512; i += 256) vfrag[i] = (_Float16)0.0f;
  __syncthreads();

  // ---- embedding: tokens = inputs_b (32x32) @ emb_W (32x64) + emb_b ----
  const float* inp = inputs + (size_t)b * (TIN * TOK);
  for (int t = wave; t < 2 * 4; t += 8) {
    const int m0 = (t >> 2) * 16, n0 = (t & 3) * 16;
    v16h a  = load_a_gf32(inp, TOK, m0, 0);
    v16h bb = load_frag(ws + OFF_EMB, n0 >> 4);
    v8f  c  = {};
    c = wmma16(a, bb, c);
    store_c_emb(c, xbuf, xfrag, m0, n0, emb_b);
  }
  if (tid < EMBD) {
    const float hval = hidden[(size_t)b * EMBD + tid];
    xbuf[32 * EMBD + tid] = hval;
    xfrag[afrag_idx(32, tid, 2)] = (_Float16)hval;
  }
  __syncthreads();

  // ---- transformer layers ----
  for (int layer = 0; layer < NDEP; ++layer) {
    const _Float16* wsq = ws + OFF_WQ + layer * SZ_BIG;
    const _Float16* wsk = ws + OFF_WK + layer * SZ_BIG;
    const _Float16* wsv = ws + OFF_WV + layer * SZ_BIG;
    const _Float16* wsu = ws + OFF_WU + layer * SZ_BIG;
    const _Float16* ws1 = ws + OFF_W1 + layer * SZ_BIG;
    const _Float16* ws2 = ws + OFF_W2 + layer * SZ_BIG;
    const float* bul = bu + layer * EMBD;
    const float* g1  = ln1_g + layer * EMBD;
    const float* b1  = ln1_b + layer * EMBD;
    const float* fb1 = ffb1 + layer * FFD;
    const float* fb2 = ffb2 + layer * EMBD;
    const float* g2  = ln2_g + layer * EMBD;
    const float* b2  = ln2_b + layer * EMBD;

    if (tid < 8) {  // warm the layer's weight streams into the cache hierarchy
      __builtin_prefetch(wsq + tid * 4096, 0, 1);
      __builtin_prefetch(ws1 + tid * 4096, 0, 1);
    }

    // QKV: (48x64) @ (64x256); scale folded into q,k stores
    for (int t = wave; t < 3 * 48; t += 8) {
      const int which = t / 48, tt = t % 48;
      const int m0 = (tt >> 4) * 16, n0 = (tt & 15) * 16;
      const _Float16* W = (which == 0) ? wsq : (which == 1) ? wsk : wsv;
      const float mul = (which == 2) ? 1.0f : scale;
      v8f c = {};
#pragma unroll
      for (int kt = 0; kt < 2; ++kt) {
        v16h a  = load_frag(xfrag, (m0 >> 4) * 2 + kt);
        v16h bb = load_frag(W, kt * 16 + (n0 >> 4));
        c = wmma16(a, bb, c);
      }
      store_qkv(c, qfrag, kfrag, vfrag, which, m0, n0, mul);
    }
    __syncthreads();

    // attention per head
    for (int h = 0; h < HEADS; ++h) {
      // scores = q_h (48x64) @ k_h^T (64x48) -> sbuf f32
      for (int t = wave; t < 9; t += 8) {
        const int mt = t / 3, nt = t % 3;
        v8f c = {};
#pragma unroll
        for (int kt = 0; kt < 2; ++kt) {
          v16h a  = load_frag(qfrag, h * 6 + mt * 2 + kt);
          v16h bb = load_frag(kfrag, h * 6 + kt * 3 + nt);
          c = wmma16(a, bb, c);
        }
        store_c_f32(c, sbuf, EMBD, mt * 16, nt * 16);
      }
      __syncthreads();
      // softmax over k=0..32 -> afrag (padded positions stay exact zeros)
      for (int r = wave; r < TSEQ; r += 8) {
        const float v0 = sbuf[r * EMBD + lane];
        const float v1 = (lane == 0) ? sbuf[r * EMBD + 32] : -3.0e38f;
        const float mx = wave_max(fmaxf(v0, v1));
        const float e0 = __expf(v0 - mx);
        const float e1 = (lane == 0) ? __expf(v1 - mx) : 0.0f;
        const float inv = 1.0f / wave_sum(e0 + e1);
        afrag[afrag_idx(r, lane, 2)] = (_Float16)(e0 * inv);
        if (lane == 0) afrag[afrag_idx(r, 32, 2)] = (_Float16)(e1 * inv);
      }
      __syncthreads();
      // out_h = att (48x64, K padded) @ v_h -> ofrag cols [h*64, h*64+64)
      for (int t = wave; t < 12; t += 8) {
        const int m0 = (t >> 2) * 16, n0 = (t & 3) * 16;
        v8f c = {};
#pragma unroll
        for (int kt = 0; kt < 2; ++kt) {
          v16h a  = load_frag(afrag, (m0 >> 4) * 2 + kt);
          v16h bb = load_frag(vfrag, h * 8 + kt * 4 + (n0 >> 4));
          c = wmma16(a, bb, c);
        }
        store_c_afrag(c, ofrag, 8, m0, h * EMBD + n0);
      }
      __syncthreads();
    }

    // a = out (48x256) @ Wu (256x64) + bu + x -> sbuf
    for (int t = wave; t < 12; t += 8) {
      const int m0 = (t >> 2) * 16, n0 = (t & 3) * 16;
      v8f c = {};
#pragma unroll
      for (int kt = 0; kt < 8; ++kt) {
        v16h a  = load_frag(ofrag, (m0 >> 4) * 8 + kt);
        v16h bb = load_frag(wsu, kt * 4 + (n0 >> 4));
        c = wmma16(a, bb, c);
      }
      store_c_f32_res(c, sbuf, EMBD, m0, n0, bul, xbuf);
    }
    __syncthreads();
    // LN1 -> xbuf (f32) + xfrag (f16 A-fragments)
    for (int r = wave; r < TSEQ; r += 8) {
      const float a0 = sbuf[r * EMBD + lane];
      const float a1 = sbuf[r * EMBD + 32 + lane];
      const float mu = wave_sum(a0 + a1) * (1.0f / 64.0f);
      const float d0 = a0 - mu, d1 = a1 - mu;
      const float var = wave_sum(d0 * d0 + d1 * d1) * (1.0f / 64.0f);
      const float rs = rsqrtf(var + 1e-5f);
      const float o0 = d0 * rs * g1[lane] + b1[lane];
      const float o1 = d1 * rs * g1[32 + lane] + b1[32 + lane];
      xbuf[r * EMBD + lane] = o0;
      xbuf[r * EMBD + 32 + lane] = o1;
      xfrag[afrag_idx(r, lane, 2)] = (_Float16)o0;
      xfrag[afrag_idx(r, 32 + lane, 2)] = (_Float16)o1;
    }
    __syncthreads();

    // ff hidden = relu(x @ W1 + b1) -> ofrag
    for (int t = wave; t < 48; t += 8) {
      const int m0 = (t >> 4) * 16, n0 = (t & 15) * 16;
      v8f c = {};
#pragma unroll
      for (int kt = 0; kt < 2; ++kt) {
        v16h a  = load_frag(xfrag, (m0 >> 4) * 2 + kt);
        v16h bb = load_frag(ws1, kt * 16 + (n0 >> 4));
        c = wmma16(a, bb, c);
      }
      store_c_afrag_relu(c, ofrag, 8, m0, n0, fb1);
    }
    __syncthreads();
    // f = hidden @ W2 + b2 + x -> sbuf
    for (int t = wave; t < 12; t += 8) {
      const int m0 = (t >> 2) * 16, n0 = (t & 3) * 16;
      v8f c = {};
#pragma unroll
      for (int kt = 0; kt < 8; ++kt) {
        v16h a  = load_frag(ofrag, (m0 >> 4) * 8 + kt);
        v16h bb = load_frag(ws2, kt * 4 + (n0 >> 4));
        c = wmma16(a, bb, c);
      }
      store_c_f32_res(c, sbuf, EMBD, m0, n0, fb2, xbuf);
    }
    __syncthreads();
    // LN2 -> xbuf + xfrag
    for (int r = wave; r < TSEQ; r += 8) {
      const float a0 = sbuf[r * EMBD + lane];
      const float a1 = sbuf[r * EMBD + 32 + lane];
      const float mu = wave_sum(a0 + a1) * (1.0f / 64.0f);
      const float d0 = a0 - mu, d1 = a1 - mu;
      const float var = wave_sum(d0 * d0 + d1 * d1) * (1.0f / 64.0f);
      const float rs = rsqrtf(var + 1e-5f);
      const float o0 = d0 * rs * g2[lane] + b2[lane];
      const float o1 = d1 * rs * g2[32 + lane] + b2[32 + lane];
      xbuf[r * EMBD + lane] = o0;
      xbuf[r * EMBD + 32 + lane] = o1;
      xfrag[afrag_idx(r, lane, 2)] = (_Float16)o0;
      xfrag[afrag_idx(r, 32 + lane, 2)] = (_Float16)o1;
    }
    __syncthreads();
  }

  // outputs = x @ toprobs_W + toprobs_b -> sbuf (rows 0..32 valid)
  for (int t = wave; t < 12; t += 8) {
    const int m0 = (t >> 2) * 16, n0 = (t & 3) * 16;
    v8f c = {};
#pragma unroll
    for (int kt = 0; kt < 2; ++kt) {
      v16h a  = load_frag(xfrag, (m0 >> 4) * 2 + kt);
      v16h bb = load_frag(ws + OFF_TOP, kt * 4 + (n0 >> 4));
      c = wmma16(a, bb, c);
    }
    store_c_f32_bias(c, sbuf, EMBD, m0, n0, topb);
  }
  __syncthreads();

  // ---- classifier + q heads (matrix-vector, plain f32 VALU) ----
  // scratch lives in sbuf rows >= 40 (outputs rows > 32 are unused)
  const float* hv  = sbuf + 32 * EMBD;  // outputs[:, -1, :]
  const float* phb = p_h + (size_t)b * 32;
  float* z1    = sbuf + 2560;  // 128
  float* z2    = sbuf + 2688;  // 64
  float* probs = sbuf + 2752;  // 8
  float* pr    = sbuf + 2760;  // 32

  if (tid < 128) {
    float s = cb1[tid];
    for (int j = 0; j < 64; ++j) s += hv[j] * cW1[j * 128 + tid];
    for (int j = 0; j < 32; ++j) s += phb[j] * cW1[(64 + j) * 128 + tid];
    z1[tid] = fmaxf(s, 0.0f);
  }
  __syncthreads();
  if (tid < 64) {
    float s = cb2[tid];
    for (int j = 0; j < 128; ++j) s += z1[j] * cW2[j * 64 + tid];
    z2[tid] = fmaxf(s, 0.0f);
  }
  __syncthreads();
  if (tid < 8) {
    float s = cb3[tid];
    for (int j = 0; j < 64; ++j) s += z2[j] * cW3[j * 8 + tid];
    probs[tid] = s;
  }
  __syncthreads();
  if (tid == 0) {
    float mx = -3.0e38f;
    for (int j = 0; j < 8; ++j) mx = fmaxf(mx, probs[j]);
    float s = 0.0f;
    for (int j = 0; j < 8; ++j) { const float e = __expf(probs[j] - mx); probs[j] = e; s += e; }
    const float inv = 1.0f / s;
    for (int j = 0; j < 8; ++j) probs[j] *= inv;
  }
  __syncthreads();

  const size_t hoff  = (size_t)Bn * QOUT;
  const size_t poff1 = hoff + (size_t)Bn * EMBD;
  const size_t poff2 = poff1 + (size_t)Bn * 32;

  if (tid < 32) {
    float s = 0.0f;
    for (int j = 0; j < 8; ++j) s += probs[j] * pew[j * 32 + tid];
    pr[tid] = s;
    out[poff1 + (size_t)b * 32 + tid] = s;
    out[poff2 + (size_t)b * 32 + tid] = s;
  }
  if (tid < EMBD) out[hoff + (size_t)b * EMBD + tid] = hv[tid];
  __syncthreads();

  if (tid < 6) {  // q_basic = [outputs[:,0,:], p_rep] @ qself_W + qself_b
    float s = qsb[tid];
    for (int j = 0; j < 64; ++j) s += sbuf[j] * qsW[j * 6 + tid];
    for (int j = 0; j < 32; ++j) s += pr[j] * qsW[(64 + j) * 6 + tid];
    out[(size_t)b * QOUT + tid] = s;
  }
  if (tid < 20) {  // q_en over enemies (outputs rows 1..20)
    float s = qib[0];
    for (int j = 0; j < 32; ++j) s += pr[j] * qiW[64 + j];
    for (int j = 0; j < 64; ++j) s += sbuf[(1 + tid) * EMBD + j] * qiW[j];
    out[(size_t)b * QOUT + 6 + tid] = s;
  }
}

extern "C" void kernel_launch(void* const* d_in, const int* in_sizes, int n_in,
                              void* d_out, int out_size, void* d_ws, size_t ws_size,
                              hipStream_t stream) {
  const float* inputs = (const float*)d_in[0];
  const float* hidden = (const float*)d_in[1];
  const float* p_h    = (const float*)d_in[2];
  const float* emb_W  = (const float*)d_in[3];
  const float* emb_b  = (const float*)d_in[4];
  const float* Wq     = (const float*)d_in[5];
  const float* Wk     = (const float*)d_in[6];
  const float* Wv     = (const float*)d_in[7];
  const float* Wu     = (const float*)d_in[8];
  const float* bu     = (const float*)d_in[9];
  const float* ln1_g  = (const float*)d_in[10];
  const float* ln1_b  = (const float*)d_in[11];
  const float* ffW1   = (const float*)d_in[12];
  const float* ffb1   = (const float*)d_in[13];
  const float* ffW2   = (const float*)d_in[14];
  const float* ffb2   = (const float*)d_in[15];
  const float* ln2_g  = (const float*)d_in[16];
  const float* ln2_b  = (const float*)d_in[17];
  const float* topW   = (const float*)d_in[18];
  const float* topb   = (const float*)d_in[19];
  const float* cW1    = (const float*)d_in[20];
  const float* cb1    = (const float*)d_in[21];
  const float* cW2    = (const float*)d_in[22];
  const float* cb2    = (const float*)d_in[23];
  const float* cW3    = (const float*)d_in[24];
  const float* cb3    = (const float*)d_in[25];
  const float* qsW    = (const float*)d_in[26];
  const float* qsb    = (const float*)d_in[27];
  const float* qiW    = (const float*)d_in[28];
  const float* qib    = (const float*)d_in[29];
  const float* pew    = (const float*)d_in[30];

  _Float16* wsh = (_Float16*)d_ws;
  const int Bn = in_sizes[0] / (TIN * TOK);

  // ---- prep: swizzle all GEMM weights into f16 B-fragment layout ----
  swizzle_weight<<<(32 * 64 + 255) / 256, 256, 0, stream>>>(emb_W, wsh + OFF_EMB, 32, 64);
  for (int l = 0; l < NDEP; ++l) {
    swizzle_weight<<<64, 256, 0, stream>>>(Wq + l * 64 * 256, wsh + OFF_WQ + l * SZ_BIG, 64, 256);
    swizzle_weight<<<64, 256, 0, stream>>>(Wk + l * 64 * 256, wsh + OFF_WK + l * SZ_BIG, 64, 256);
    swizzle_weight<<<64, 256, 0, stream>>>(Wv + l * 64 * 256, wsh + OFF_WV + l * SZ_BIG, 64, 256);
    swizzle_weight<<<64, 256, 0, stream>>>(Wu + l * 256 * 64, wsh + OFF_WU + l * SZ_BIG, 256, 64);
    swizzle_weight<<<64, 256, 0, stream>>>(ffW1 + l * 64 * 256, wsh + OFF_W1 + l * SZ_BIG, 64, 256);
    swizzle_weight<<<64, 256, 0, stream>>>(ffW2 + l * 256 * 64, wsh + OFF_W2 + l * SZ_BIG, 256, 64);
  }
  swizzle_weight<<<16, 256, 0, stream>>>(topW, wsh + OFF_TOP, 64, 64);

  phase_updet_wmma<<<Bn, 256, 0, stream>>>(
      inputs, hidden, p_h, emb_b, bu, ln1_g, ln1_b, ffb1, ffb2, ln2_g, ln2_b,
      topb, cW1, cb1, cW2, cb2, cW3, cb3, qsW, qsb, qiW, qib, pew,
      (const _Float16*)d_ws, (float*)d_out, Bn);
}
